// SelfAttention_23553600651604
// MI455X (gfx1250) — compile-verified, matrix operand order checked by reading, not run on hardware
//
#include <hip/hip_runtime.h>
#include <hip/hip_bf16.h>

#define SEQ 2048
#define NB  2
#define EMB 1024
#define NH  16
#define HD  64

typedef __attribute__((ext_vector_type(16))) _Float16 v16h;
typedef __attribute__((ext_vector_type(8)))  float    v8f;

__device__ __forceinline__ v8f wmma16x16x32(v16h a, v16h b, v8f c) {
  // D = A(16x32 f16) x B(32x16 f16) + C(16x16 f32)
  return __builtin_amdgcn_wmma_f32_16x16x32_f16(false, a, false, b, (short)0, c,
                                                false, false);
}

// A-matrix fragment (16x32 f16), ISA 7.12.2 layout. Per lane the k indices are
// two contiguous 8-half (16B) runs -> vectorizes to 2x ds_load_b128.
__device__ __forceinline__ v16h frag_a(const _Float16* base, int stride) {
  const int lane = threadIdx.x & 31;
  const int m = lane & 15, lh = lane >> 4;
  v16h r;
#pragma unroll
  for (int i = 0; i < 16; ++i) {
    int k = ((i & 8) << 1) + (lh << 3) + (i & 7);
    r[i] = base[m * stride + k];
  }
  return r;
}

// B-matrix fragment B[k][n] from LDS stored [n][k]; same 16B-run property.
__device__ __forceinline__ v16h frag_b_transposed(const _Float16* base, int stride) {
  const int lane = threadIdx.x & 31;
  const int nn = lane & 15, lh = lane >> 4;
  v16h r;
#pragma unroll
  for (int i = 0; i < 16; ++i) {
    int k = ((i & 8) << 1) + (lh << 3) + (i & 7);
    r[i] = base[nn * stride + k];
  }
  return r;
}

// Row reductions over the 16 lanes of a wave32 half (masks < 16 stay in-group)
__device__ __forceinline__ float redmax16(float v) {
  v = fmaxf(v, __shfl_xor(v, 1));
  v = fmaxf(v, __shfl_xor(v, 2));
  v = fmaxf(v, __shfl_xor(v, 4));
  v = fmaxf(v, __shfl_xor(v, 8));
  return v;
}
__device__ __forceinline__ float redsum16(float v) {
  v += __shfl_xor(v, 1);
  v += __shfl_xor(v, 2);
  v += __shfl_xor(v, 4);
  v += __shfl_xor(v, 8);
  return v;
}

// --- CDNA5 async global->LDS copy (ASYNCcnt-tracked), §15.18 opcode 98 ------
__device__ __forceinline__ unsigned lds_addr(const void* p) {
  // flat LDS addresses: low 32 bits are the wave-relative LDS offset
  return (unsigned)(unsigned long long)p;
}
__device__ __forceinline__ void async_copy_b128(unsigned lds_off, const void* g) {
  asm volatile("global_load_async_to_lds_b128 %0, %1, off"
               :: "v"(lds_off), "v"(g) : "memory");
}
__device__ __forceinline__ void wait_async0() {
  asm volatile("s_wait_asynccnt 0x0" ::: "memory");
}

// ---------------------------------------------------------------------------
// Kernel 0: one-shot W f32 -> f16 conversion (4 MB read, 2 MB write, once).
// ---------------------------------------------------------------------------
__global__ __launch_bounds__(256) void cvt_w_kernel(const float* __restrict__ W,
                                                    _Float16* __restrict__ Wh) {
  long i = ((long)blockIdx.x * 256 + threadIdx.x) * 4;
  float4 w = *(const float4*)(W + i);
  Wh[i + 0] = (_Float16)w.x; Wh[i + 1] = (_Float16)w.y;
  Wh[i + 2] = (_Float16)w.z; Wh[i + 3] = (_Float16)w.w;
}

// ---------------------------------------------------------------------------
// Kernel 1: fused flash attention. Block = 256 thr = 8 waves; each wave owns a
// 16-query tile of one (n,h); single pass over K in blocks of 64 keys.
// Emits ctx[N,S,H*D] as f16 (consumed as f16 by the FC GEMM anyway).
// ---------------------------------------------------------------------------
__global__ __launch_bounds__(256) void attn_kernel(
    const float* __restrict__ query, const float* __restrict__ keys,
    const float* __restrict__ values, const int* __restrict__ mask,
    _Float16* __restrict__ ctx) {
  __shared__ alignas(16) _Float16 kLDS[64][72];      // [key][d]
  __shared__ alignas(16) _Float16 vLDS[64][72];      // [d][key]  (transposed!)
  __shared__ alignas(16) _Float16 pLDS[8][16][72];   // per-wave P relayout

  const int tid  = threadIdx.x;
  const int wid  = tid >> 5;
  const int lane = tid & 31;
  const int lh   = lane >> 4;
  const int ln   = lane & 15;
  const int qt = blockIdx.x, h = blockIdx.y, n = blockIdx.z;
  const int q0 = qt * 128 + wid * 16;

  const float* qp = query  + ((long)n * SEQ + q0) * EMB + h * HD;
  const float* kp = keys   + (long)n * SEQ * EMB + h * HD;
  const float* vp = values + (long)n * SEQ * EMB + h * HD;
  const int*   mp = mask   + (long)n * SEQ * SEQ;

  // Q fragments (two K=32 chunks over D=64); softmax 1/sqrt(EMB)=1/32 folded in
  v16h qa[2];
#pragma unroll
  for (int c = 0; c < 2; ++c)
#pragma unroll
    for (int i = 0; i < 16; ++i) {
      int k = ((i & 8) << 1) + (lh << 3) + (i & 7) + c * 32;
      qa[c][i] = (_Float16)(qp[(long)ln * EMB + k] * 0.03125f);
    }

  float mrow[8], lrow[8];
  v8f o[4];
#pragma unroll
  for (int r = 0; r < 8; ++r) { mrow[r] = -3.0e38f; lrow[r] = 0.0f; }
#pragma unroll
  for (int t = 0; t < 4; ++t)
#pragma unroll
    for (int r = 0; r < 8; ++r) o[t][r] = 0.0f;

  for (int kb = 0; kb < SEQ; kb += 64) {
    // Stage K (row-major) and V (transposed) tiles, f32 -> f16
#pragma unroll
    for (int it = 0; it < 4; ++it) {
      int idx = tid + it * 256;
      int row = idx >> 4;               // key 0..63
      int c4  = (idx & 15) << 2;        // d 0..60
      float4 kv = *(const float4*)(kp + (long)(kb + row) * EMB + c4);
      float4 vv = *(const float4*)(vp + (long)(kb + row) * EMB + c4);
      kLDS[row][c4 + 0] = (_Float16)kv.x; kLDS[row][c4 + 1] = (_Float16)kv.y;
      kLDS[row][c4 + 2] = (_Float16)kv.z; kLDS[row][c4 + 3] = (_Float16)kv.w;
      vLDS[c4 + 0][row] = (_Float16)vv.x; vLDS[c4 + 1][row] = (_Float16)vv.y;
      vLDS[c4 + 2][row] = (_Float16)vv.z; vLDS[c4 + 3][row] = (_Float16)vv.w;
    }
    if (kb + 64 < SEQ) {                 // global_prefetch_b8 of the next tiles
      __builtin_prefetch(kp + (long)(kb + 64 + (tid & 63)) * EMB, 0, 0);
      __builtin_prefetch(vp + (long)(kb + 64 + (tid & 63)) * EMB, 0, 0);
    }
    __syncthreads();

    // S = (Q/32) * K^T for four 16-key sub-tiles; mask -> -1e30
    float sv[4][8];
#pragma unroll
    for (int t = 0; t < 4; ++t) {
      v8f s;
#pragma unroll
      for (int r = 0; r < 8; ++r) s[r] = 0.0f;
      v16h b0 = frag_b_transposed(&kLDS[t * 16][0], 72);
      v16h b1 = frag_b_transposed(&kLDS[t * 16][32], 72);
      s = wmma16x16x32(qa[0], b0, s);
      s = wmma16x16x32(qa[1], b1, s);
#pragma unroll
      for (int r = 0; r < 8; ++r) {
        int row = q0 + r + 8 * lh;
        int mv  = mp[(long)row * SEQ + kb + t * 16 + ln];
        sv[t][r] = mv ? s[r] : -1.0e30f;
      }
    }

    // Online softmax: fused reduction trees (one max, one sum per row)
#pragma unroll
    for (int r = 0; r < 8; ++r) {
      float mloc = fmaxf(fmaxf(sv[0][r], sv[1][r]), fmaxf(sv[2][r], sv[3][r]));
      float mnew = fmaxf(mrow[r], redmax16(mloc));
      float corr = __expf(mrow[r] - mnew);
      mrow[r] = mnew;
      float psum = 0.0f;
#pragma unroll
      for (int t = 0; t < 4; ++t) {
        float p = __expf(sv[t][r] - mnew);
        sv[t][r] = p;
        psum += p;
      }
      lrow[r] = lrow[r] * corr + redsum16(psum);
#pragma unroll
      for (int t = 0; t < 4; ++t) o[t][r] *= corr;
#pragma unroll
      for (int t = 0; t < 4; ++t)
        pLDS[wid][r + 8 * lh][t * 16 + ln] = (_Float16)sv[t][r];
    }

    // O += P(16x64) * V(64x64); all fragments are contiguous-run b128 loads
    v16h pa0 = frag_a(&pLDS[wid][0][0], 72);
    v16h pa1 = frag_a(&pLDS[wid][0][32], 72);
#pragma unroll
    for (int ct = 0; ct < 4; ++ct) {
      o[ct] = wmma16x16x32(pa0, frag_b_transposed(&vLDS[ct * 16][0], 72), o[ct]);
      o[ct] = wmma16x16x32(pa1, frag_b_transposed(&vLDS[ct * 16][32], 72), o[ct]);
    }
    __syncthreads();
  }

  // Normalize and write f16 ctx[n, q, h*64 + d]
#pragma unroll
  for (int r = 0; r < 8; ++r) {
    float inv = 1.0f / lrow[r];
    _Float16* op = ctx + ((long)n * SEQ + q0 + r + 8 * lh) * EMB + h * HD;
#pragma unroll
    for (int ct = 0; ct < 4; ++ct)
      op[ct * 16 + ln] = (_Float16)(o[ct][r] * inv);
  }
}

// ---------------------------------------------------------------------------
// Kernel 2: out[4096, E] = ctx_h @ Wh^T + b. Both operands already f16 in
// memory, so tiles are staged with CDNA5 async global->LDS copies (ASYNCcnt),
// double-buffered, one barrier per K-step. 8 waves, 2x2 WMMA tiles each.
// ---------------------------------------------------------------------------
__global__ __launch_bounds__(256) void fc_kernel(
    const _Float16* __restrict__ ctx, const _Float16* __restrict__ Wh,
    const float* __restrict__ bias, float* __restrict__ out) {
  __shared__ alignas(16) _Float16 aLDS[2][128][40];
  __shared__ alignas(16) _Float16 bLDS[2][64][40];

  const int tid  = threadIdx.x;
  const int wid  = tid >> 5;
  const int lane = tid & 31;
  const int lh   = lane >> 4;
  const int ln   = lane & 15;
  const int n0 = blockIdx.x * 64;
  const int m0 = blockIdx.y * 128;
  const int wm = (wid >> 1) * 32;
  const int wn = (wid & 1) * 32;

  v8f acc[4];
#pragma unroll
  for (int t = 0; t < 4; ++t)
#pragma unroll
    for (int r = 0; r < 8; ++r) acc[t][r] = 0.0f;

  // Stage one 128x32 A-tile + 64x32 B-tile into LDS buffer `buf` (raw f16,
  // 16B chunks; row pad 40 halves keeps every chunk 16B aligned).
  auto stage = [&](int buf, int kc) {
#pragma unroll
    for (int it = 0; it < 2; ++it) {           // A: 512 x b128
      int idx = tid + it * 256;
      int row = idx >> 2, q = (idx & 3) * 8;
      async_copy_b128(lds_addr(&aLDS[buf][row][q]),
                      ctx + (long)(m0 + row) * EMB + kc + q);
    }
    {                                          // B: 256 x b128
      int row = tid >> 2, q = (tid & 3) * 8;
      async_copy_b128(lds_addr(&bLDS[buf][row][q]),
                      Wh + (long)(n0 + row) * EMB + kc + q);
    }
  };

  stage(0, 0);
  for (int kc = 0; kc < EMB; kc += 32) {
    int buf = (kc >> 5) & 1;
    wait_async0();          // my async writes done
    __syncthreads();        // everyone's async writes done; prev reads done
    if (kc + 32 < EMB) stage(buf ^ 1, kc + 32);

    v16h a0 = frag_a(&aLDS[buf][wm][0], 40);
    v16h a1 = frag_a(&aLDS[buf][wm + 16][0], 40);
    v16h b0 = frag_b_transposed(&bLDS[buf][wn][0], 40);   // B = W^T
    v16h b1 = frag_b_transposed(&bLDS[buf][wn + 16][0], 40);
    acc[0] = wmma16x16x32(a0, b0, acc[0]);
    acc[1] = wmma16x16x32(a0, b1, acc[1]);
    acc[2] = wmma16x16x32(a1, b0, acc[2]);
    acc[3] = wmma16x16x32(a1, b1, acc[3]);
  }

#pragma unroll
  for (int ti = 0; ti < 2; ++ti)
#pragma unroll
    for (int tj = 0; tj < 2; ++tj) {
      v8f a = acc[ti * 2 + tj];
#pragma unroll
      for (int r = 0; r < 8; ++r) {
        int row = m0 + wm + ti * 16 + r + 8 * lh;
        int col = n0 + wn + tj * 16 + ln;
        out[(long)row * EMB + col] = a[r] + bias[col];
      }
    }
}

extern "C" void kernel_launch(void* const* d_in, const int* in_sizes, int n_in,
                              void* d_out, int out_size, void* d_ws, size_t ws_size,
                              hipStream_t stream) {
  (void)in_sizes; (void)n_in; (void)out_size; (void)ws_size;
  const float* values = (const float*)d_in[0];
  const float* keys   = (const float*)d_in[1];
  const float* query  = (const float*)d_in[2];
  const int*   mask   = (const int*)d_in[3];
  const float* W_fc   = (const float*)d_in[4];
  const float* b_fc   = (const float*)d_in[5];

  // workspace: ctx_h (8 MiB f16) | W_h (2 MiB f16)
  _Float16* ctx_h = (_Float16*)d_ws;
  _Float16* W_h   = (_Float16*)((char*)d_ws + (size_t)NB * SEQ * EMB * 2);

  cvt_w_kernel<<<dim3((EMB * EMB) / 1024), 256, 0, stream>>>(W_fc, W_h);
  attn_kernel<<<dim3(SEQ / 128, NH, NB), 256, 0, stream>>>(query, keys, values,
                                                           mask, ctx_h);
  fc_kernel<<<dim3(EMB / 64, (NB * SEQ) / 128), 256, 0, stream>>>(ctx_h, W_h,
                                                                  b_fc,
                                                                  (float*)d_out);
}
